// RetinaNet_51101520887812
// MI455X (gfx1250) — compile-verified, matrix operand order checked by reading, not run on hardware
//
#include <hip/hip_runtime.h>
#include <hip/hip_bf16.h>

typedef __attribute__((ext_vector_type(16))) _Float16 v16h;
typedef __attribute__((ext_vector_type(8)))  float    v8f;

#define NUM_A   49104
#define NUM_B   2
#define NUM_G   (NUM_A * NUM_B)      // 98208 flattened anchors
#define NUM_C   80
#define TOPK    1000
#define NTILE   63                   // ceil(1000/16)
#define KPAD    (NTILE * 16)         // 1008
#define CAP     2048                 // candidate buffer per class (>= K + tie slack)
#define NBIN    4096
#define IMGF    512.0f
#define INV_AREA (1.0f / (512.0f * 512.0f))
#define SCORE_THR 0.05f
#define IOU_THR   0.5f

// ---------------------------------------------------------------------------
// Kernel 1: BBoxTransform (std 0.1/0.1/0.2/0.2) + ClipBoxes -> ws boxes[G][4]
// Pure streaming, HBM-bound (~2.4 MB read, 1.6 MB write).
// ---------------------------------------------------------------------------
__global__ void retina_decode_kernel(const float* __restrict__ reg,
                                     const float* __restrict__ anchors,
                                     float* __restrict__ boxes) {
  int g = blockIdx.x * blockDim.x + threadIdx.x;
  if (g >= NUM_G) return;
  int a = g % NUM_A;
  float x1 = anchors[a * 4 + 0], y1 = anchors[a * 4 + 1];
  float x2 = anchors[a * 4 + 2], y2 = anchors[a * 4 + 3];
  float w = x2 - x1, h = y2 - y1;
  float cx = x1 + 0.5f * w, cy = y1 + 0.5f * h;
  float dx = reg[g * 4 + 0] * 0.1f, dy = reg[g * 4 + 1] * 0.1f;
  float dw = reg[g * 4 + 2] * 0.2f, dh = reg[g * 4 + 3] * 0.2f;
  float pcx = cx + dx * w, pcy = cy + dy * h;
  float pw = __expf(dw) * w, ph = __expf(dh) * h;
  boxes[g * 4 + 0] = fminf(fmaxf(pcx - 0.5f * pw, 0.0f), IMGF);
  boxes[g * 4 + 1] = fminf(fmaxf(pcy - 0.5f * ph, 0.0f), IMGF);
  boxes[g * 4 + 2] = fminf(fmaxf(pcx + 0.5f * pw, 0.0f), IMGF);
  boxes[g * 4 + 3] = fminf(fmaxf(pcy + 0.5f * ph, 0.0f), IMGF);
}

// ---------------------------------------------------------------------------
// Kernel 2: per-class threshold + top-1000, one workgroup (32 waves) / class.
// Pass 1: LDS histogram on the monotone positive-float bit pattern (DS atomics).
// Pass 2: gather candidates >= cutoff bin, bitonic-sort 2048 in LDS, emit 1000.
// All 31.4 MB of scores live in L2 (192 MB), so strided class reads are cheap.
// ---------------------------------------------------------------------------
__global__ __launch_bounds__(1024) void retina_topk_kernel(
    const float* __restrict__ cls,
    float* __restrict__ topkV,
    int*   __restrict__ topkI) {
  __shared__ unsigned int hist[NBIN];
  __shared__ float sv[CAP];
  __shared__ int   si[CAP];
  __shared__ int   s_cnt;
  __shared__ int   s_cut;

  const int c = blockIdx.x;
  const int tid = threadIdx.x;

  for (int b = tid; b < NBIN; b += 1024) hist[b] = 0u;
  if (tid == 0) s_cnt = 0;
  __syncthreads();

  for (int g = tid; g < NUM_G; g += 1024) {
    float s = cls[(size_t)g * NUM_C + c];
    if (g + 4096 < NUM_G)   // gfx1250 global_prefetch_b8, speculative
      __builtin_prefetch(&cls[(size_t)(g + 4096) * NUM_C + c], 0, 0);
    if (s > SCORE_THR) {
      unsigned bin = (__float_as_uint(s) >> 20) & (NBIN - 1);  // monotone key
      atomicAdd(&hist[bin], 1u);
    }
  }
  __syncthreads();

  if (tid == 0) {
    int cum = 0, cb = 0;
    for (int b = NBIN - 1; b >= 0; --b) {
      cum += (int)hist[b];
      if (cum >= TOPK) { cb = b; break; }
    }
    s_cut = cb;
  }
  __syncthreads();
  const int cb = s_cut;

  for (int g = tid; g < NUM_G; g += 1024) {
    float s = cls[(size_t)g * NUM_C + c];
    if (s > SCORE_THR && (int)((__float_as_uint(s) >> 20) & (NBIN - 1)) >= cb) {
      int pos = atomicAdd(&s_cnt, 1);
      if (pos < CAP) { sv[pos] = s; si[pos] = g; }
    }
  }
  __syncthreads();

  int n = s_cnt; if (n > CAP) n = CAP;
  for (int i = tid; i < CAP; i += 1024)
    if (i >= n) { sv[i] = -__builtin_inff(); si[i] = 0; }
  __syncthreads();

  // Bitonic sort, descending by score (2048 elems, 1024 threads)
  for (int k2 = 2; k2 <= CAP; k2 <<= 1) {
    for (int j = k2 >> 1; j > 0; j >>= 1) {
      for (int i = tid; i < CAP; i += 1024) {
        int ixj = i ^ j;
        if (ixj > i) {
          bool desc = ((i & k2) == 0);
          float va = sv[i], vb = sv[ixj];
          bool sw = desc ? (va < vb) : (va > vb);
          if (sw) {
            sv[i] = vb; sv[ixj] = va;
            int t = si[i]; si[i] = si[ixj]; si[ixj] = t;
          }
        }
      }
      __syncthreads();
    }
  }

  if (tid < TOPK) {
    topkV[c * TOPK + tid] = sv[tid];
    topkI[c * TOPK + tid] = si[tid];
  }
}

// ---------------------------------------------------------------------------
// Kernel 3: per-class NMS. 32 waves build the symmetric 1000x1000 suppression
// bitmatrix in LDS (128 KB, legal on the 320 KB WGP). Each 16x16 tile's
// union denominator area_i + area_j is computed as a rank-2 f16 WMMA
// (boxes normalized to [0,1] so f16 range/precision is fine; IoU is
// scale-invariant), then inter/union via VALU min/max with a divide-free
// compare. One thread then replays the exact greedy scan over bitmasks.
// ---------------------------------------------------------------------------
__global__ __launch_bounds__(1024) void retina_nms_kernel(
    const float* __restrict__ boxes,
    const float* __restrict__ topkV,
    const int*   __restrict__ topkI,
    float* __restrict__ out) {
  extern __shared__ unsigned char smem[];
  float*        sbx   = (float*)smem;                          // KPAD*4 f32
  float*        svals = sbx + KPAD * 4;                        // TOPK  f32
  _Float16*     sarea = (_Float16*)(svals + TOPK);             // KPAD  f16 (normalized areas)
  unsigned int* ssup  = (unsigned int*)(sarea + KPAD);         // TOPK*32 u32 bitmatrix
  unsigned int* skeep = ssup + TOPK * 32;                      // 32 u32

  const int c = blockIdx.x;
  const int tid = threadIdx.x;

  for (int t = tid; t < KPAD; t += 1024) {
    float b0 = 0.f, b1 = 0.f, b2 = 0.f, b3 = 0.f;
    if (t < TOPK) {
      float v = topkV[c * TOPK + t];
      int idx  = topkI[c * TOPK + t];
      b0 = boxes[idx * 4 + 0]; b1 = boxes[idx * 4 + 1];
      b2 = boxes[idx * 4 + 2]; b3 = boxes[idx * 4 + 3];
      svals[t] = v;
    }
    sbx[t * 4 + 0] = b0; sbx[t * 4 + 1] = b1;
    sbx[t * 4 + 2] = b2; sbx[t * 4 + 3] = b3;
    sarea[t] = (_Float16)(((b2 - b0) * (b3 - b1)) * INV_AREA);
  }
  for (int w = tid; w < TOPK * 32; w += 1024) ssup[w] = 0u;
  if (tid < 32) skeep[tid] = 0u;
  __syncthreads();

  // --- tiled suppression matrix, WMMA area-sum per 16x16 tile -------------
  const int wave = tid >> 5;
  const int lane = tid & 31;
  for (int tI = wave; tI < NTILE * NTILE; tI += 32) {
    int ti = tI / NTILE, tj = tI % NTILE;

    // A(16x32): col0 = area_i, col1 = 1 ; B(32x16): row0 = 1, row1 = area_j
    v16h aF, bF;
#pragma unroll
    for (int e = 0; e < 16; ++e) { aF[e] = (_Float16)0.0f; bF[e] = (_Float16)0.0f; }
    if (lane < 16) {
      aF[0] = sarea[ti * 16 + lane]; aF[1] = (_Float16)1.0f;
      bF[0] = (_Float16)1.0f;        bF[1] = sarea[tj * 16 + lane];
    }
    v8f acc;
#pragma unroll
    for (int e = 0; e < 8; ++e) acc[e] = 0.0f;
    acc = __builtin_amdgcn_wmma_f32_16x16x32_f16(
        /*neg_a=*/false, aF, /*neg_b=*/false, bF,
        /*c_mod=*/(short)0, acc, /*reuse_a=*/false, /*reuse_b=*/false);

    // D layout: lanes 0-15 -> M=r, lanes 16-31 -> M=8+r ; N = lane&15
    const int i0 = ti * 16 + ((lane >= 16) ? 8 : 0);
    const int j  = tj * 16 + (lane & 15);
#pragma unroll
    for (int r = 0; r < 8; ++r) {
      int i = i0 + r;
      if (i < TOPK && j < TOPK && i != j) {
        float xx1 = fmaxf(sbx[i * 4 + 0], sbx[j * 4 + 0]);
        float yy1 = fmaxf(sbx[i * 4 + 1], sbx[j * 4 + 1]);
        float xx2 = fminf(sbx[i * 4 + 2], sbx[j * 4 + 2]);
        float yy2 = fminf(sbx[i * 4 + 3], sbx[j * 4 + 3]);
        float inter = fmaxf(xx2 - xx1, 0.0f) * fmaxf(yy2 - yy1, 0.0f) * INV_AREA;
        float uni = acc[r] - inter + 1e-9f;   // (area_i + area_j) from WMMA
        if (inter > IOU_THR * uni) {          // iou > 0.5, divide-free
          atomicOr(&ssup[i * 32 + (j >> 5)], 1u << (j & 31));
        }
      }
    }
  }
  __syncthreads();

  // --- exact greedy scan over the symmetric bitmatrix ---------------------
  if (tid == 0) {
    for (int i = 0; i < TOPK; ++i) {
      unsigned any = 0u;
      const unsigned* row = &ssup[i * 32];
      for (int w = 0; w < 32; ++w) any |= (skeep[w] & row[w]);
      if (any == 0u && svals[i] > SCORE_THR)
        skeep[i >> 5] |= (1u << (i & 31));
    }
  }
  __syncthreads();

  // --- outputs: scores | labels | boxes(*keep) | keep ---------------------
  const int NQ = NUM_C * TOPK;   // 80000
  for (int t = tid; t < TOPK; t += 1024) {
    int q = c * TOPK + t;
    bool kept = ((skeep[t >> 5] >> (t & 31)) & 1u) != 0u;
    float m = kept ? 1.0f : 0.0f;
    out[q]            = kept ? svals[t] : 0.0f;
    out[NQ + q]       = (float)c;
    out[2 * NQ + q * 4 + 0] = sbx[t * 4 + 0] * m;
    out[2 * NQ + q * 4 + 1] = sbx[t * 4 + 1] * m;
    out[2 * NQ + q * 4 + 2] = sbx[t * 4 + 2] * m;
    out[2 * NQ + q * 4 + 3] = sbx[t * 4 + 3] * m;
    out[6 * NQ + q]   = m;
  }
}

// ---------------------------------------------------------------------------
extern "C" void kernel_launch(void* const* d_in, const int* in_sizes, int n_in,
                              void* d_out, int out_size, void* d_ws, size_t ws_size,
                              hipStream_t stream) {
  const float* cls  = (const float*)d_in[0];  // [2,49104,80]
  const float* reg  = (const float*)d_in[1];  // [2,49104,4]
  const float* anch = (const float*)d_in[2];  // [49104,4]
  float* out = (float*)d_out;                 // 560000 f32

  // workspace layout: boxes[G*4] | topkV[C*K] | topkI[C*K]  (~2.2 MB)
  float* boxes = (float*)d_ws;
  float* topkV = boxes + (size_t)NUM_G * 4;
  int*   topkI = (int*)(topkV + (size_t)NUM_C * TOPK);

  retina_decode_kernel<<<(NUM_G + 255) / 256, 256, 0, stream>>>(reg, anch, boxes);
  retina_topk_kernel<<<NUM_C, 1024, 0, stream>>>(cls, topkV, topkI);

  size_t smem = (size_t)(KPAD * 4 + TOPK) * sizeof(float)   // boxes + vals
              + (size_t)KPAD * sizeof(_Float16)             // f16 areas
              + (size_t)TOPK * 32 * sizeof(unsigned int)    // suppression bits
              + 32 * sizeof(unsigned int);                  // keep words (~150 KB)
  retina_nms_kernel<<<NUM_C, 1024, smem, stream>>>(boxes, topkV, topkI, out);
}
